// LightHeadRCNNResNet101_Head_27453430956790
// MI455X (gfx1250) — compile-verified, matrix operand order checked by reading, not run on hardware
//
#include <hip/hip_runtime.h>
#include <hip/hip_bf16.h>

// ---------------------------------------------------------------------------
// CDNA5 (gfx1250) implementation of LightHeadRCNN head:
//   conv1(15x1 / 1x15, Cin=2048, Cout=256)  -> bf16 WMMA GEMM, tap-shifted B
//   conv2(1x15 / 15x1, Cin=256,  Cout=490)  -> bf16 WMMA GEMM, tap-shifted B
//   h = relu(col + row); PSROI-align max pool (1000 ROIs) -> (1000,490)
//   fc1 (2048x490) + relu; heads (324+81 x 2048) -> d_out
// All matrix math uses v_wmma_f32_16x16x32_bf16 (f32 accumulate).
// Wave tile 32x64 (8 WMMAs / K-step), block tile 128x128 (8 wave32 waves).
// ---------------------------------------------------------------------------

typedef __attribute__((ext_vector_type(16))) __bf16 v16bf;
typedef __attribute__((ext_vector_type(8)))  __bf16 v8bf;
typedef __attribute__((ext_vector_type(8)))  float  v8f;

union FragU {
    struct { v8bf lo; v8bf hi; } p;
    v16bf f;
};

// ---------------------------------------------------------------------------
// Generic bf16 WMMA GEMM:  C[M x N] = sum_{tap,c} A[tap][m][c] * B[spat(n,tap)][c]
//   A: tap-major repacked weights, row stride Kc (padded channels)
//   B: spatial-major activations, [spatial][Kc] bf16
//   spat(n,tap) = (n / nDiv) * bRowStride + (n % nDiv) + bColBase + tap*tapStride
// ---------------------------------------------------------------------------
struct GemmP {
    const __bf16* A;
    const __bf16* B;
    const float*  bias;
    int M, N, Mp, Kc, taps, tapStride, bRowStride, bColBase, nDiv;
    // MODE 0: fp32 plain store
    float* outF; int ldF;
    // MODE 1: bf16 store into padded spatial-major layout (conv intermediate)
    // MODE 2: bf16 relu store, [n][m] layout (fc1 activations)
    __bf16* outH; int oRowStride, oBase, oCp;
    // MODE 3: head scatter into d_out (loc rows then score rows)
    float* out0; const float* bias0; int M0;
    float* out1; const float* bias1;
};

template<int MODE>
__global__ __launch_bounds__(256)
void gemm_bf16_wmma(GemmP p) {
    const int lane  = threadIdx.x & 31;
    const int wid   = threadIdx.x >> 5;
    const int l15   = lane & 15;
    const int half8 = (lane >> 4) << 3;              // 0 for lanes 0-15, 8 for 16-31
    const int m0w   = blockIdx.x * 128 + (wid & 3) * 32;
    const int n0w   = blockIdx.y * 128 + (wid >> 2) * 64;

    // Per-lane spatial index of the B column for each of the four N tiles.
    int spat[4];
#pragma unroll
    for (int nt = 0; nt < 4; ++nt) {
        int n = n0w + nt * 16 + l15;
        spat[nt] = (n / p.nDiv) * p.bRowStride + (n % p.nDiv) + p.bColBase;
    }

    const long Kc = (long)p.Kc;
    const __bf16* Abase0 = p.A + (long)(m0w + l15) * Kc + half8;
    const __bf16* Abase1 = p.A + (long)(m0w + 16 + l15) * Kc + half8;
    const long AtapStride = (long)p.Mp * Kc;

    v8f acc[2][4];
#pragma unroll
    for (int mt = 0; mt < 2; ++mt)
#pragma unroll
        for (int nt = 0; nt < 4; ++nt)
            acc[mt][nt] = (v8f){};

    for (int tap = 0; tap < p.taps; ++tap) {
        const __bf16* a0p = Abase0 + (long)tap * AtapStride;
        const __bf16* a1p = Abase1 + (long)tap * AtapStride;
        const __bf16* bp[4];
#pragma unroll
        for (int nt = 0; nt < 4; ++nt)
            bp[nt] = p.B + (long)(spat[nt] + tap * p.tapStride) * Kc + half8;

        for (int c0 = 0; c0 < p.Kc; c0 += 32) {
            FragU a0, a1, b[4];
            a0.p.lo = *(const v8bf*)(a0p + c0);
            a0.p.hi = *(const v8bf*)(a0p + c0 + 16);
            a1.p.lo = *(const v8bf*)(a1p + c0);
            a1.p.hi = *(const v8bf*)(a1p + c0 + 16);
#pragma unroll
            for (int nt = 0; nt < 4; ++nt) {
                b[nt].p.lo = *(const v8bf*)(bp[nt] + c0);
                b[nt].p.hi = *(const v8bf*)(bp[nt] + c0 + 16);
            }
            __builtin_prefetch(bp[0] + c0 + 256, 0, 3);   // global_prefetch_b8
            __builtin_prefetch(bp[2] + c0 + 256, 0, 3);
#pragma unroll
            for (int nt = 0; nt < 4; ++nt) {
                acc[0][nt] = __builtin_amdgcn_wmma_f32_16x16x32_bf16(false, a0.f, false, b[nt].f, (short)0, acc[0][nt], false, false);
                acc[1][nt] = __builtin_amdgcn_wmma_f32_16x16x32_bf16(false, a1.f, false, b[nt].f, (short)0, acc[1][nt], false, false);
            }
        }
    }

#pragma unroll
    for (int mt = 0; mt < 2; ++mt) {
#pragma unroll
        for (int nt = 0; nt < 4; ++nt) {
            v8f a = acc[mt][nt];
#pragma unroll
            for (int r = 0; r < 8; ++r) {
                int m = m0w + mt * 16 + half8 + r;
                int n = n0w + nt * 16 + l15;
                if (m < p.M && n < p.N) {
                    float v = a[r];
                    if (MODE == 0) {
                        p.outF[(long)m * p.ldF + n] = v + p.bias[m];
                    } else if (MODE == 1) {
                        int sp = (n / 75) * p.oRowStride + (n % 75) + p.oBase;
                        p.outH[(long)sp * p.oCp + m] = (__bf16)(v + p.bias[m]);
                    } else if (MODE == 2) {
                        float t = v + p.bias[m];
                        p.outH[(long)n * p.oCp + m] = (__bf16)(t > 0.0f ? t : 0.0f);
                    } else {
                        if (m < p.M0) p.out0[(long)n * p.M0 + m] = v + p.bias0[m];
                        else          p.out1[(long)n * 81 + (m - p.M0)] = v + p.bias1[m - p.M0];
                    }
                }
            }
        }
    }
}

// ------------------------- prep / glue kernels -----------------------------

// x (2048, 50, 75) fp32 -> xpadT[(h+7)*89 + (w+7)][c] bf16 (zero-padded border)
__global__ void k_pack_x(const float* __restrict__ x, __bf16* __restrict__ xp) {
    int i = blockIdx.x * blockDim.x + threadIdx.x;
    if (i >= 2048 * 3750) return;
    int c = i / 3750, n = i % 3750;
    int h = n / 75, w = n % 75;
    xp[(long)((h + 7) * 89 + (w + 7)) * 2048 + c] = (__bf16)x[i];
}

// w (M, K, taps) fp32 -> out[tap][m (Mp rows)][c (Kp cols)] bf16
__global__ void k_pack_w(const float* __restrict__ w, __bf16* __restrict__ out,
                         int M, int Mp, int K, int Kp, int taps) {
    int i = blockIdx.x * blockDim.x + threadIdx.x;
    if (i >= M * K * taps) return;
    int t = i % taps;
    int c = (i / taps) % K;
    int m = i / (taps * K);
    out[(long)t * Mp * Kp + (long)m * Kp + c] = (__bf16)w[i];
}

// loc_w (324,2048) + score_w (81,2048) -> packed head weights (512 x 2048) bf16
__global__ void k_pack_heads(const float* __restrict__ loc_w,
                             const float* __restrict__ score_w,
                             __bf16* __restrict__ out) {
    int i = blockIdx.x * blockDim.x + threadIdx.x;
    if (i >= 405 * 2048) return;
    int m = i / 2048, c = i % 2048;
    float v = (m < 324) ? loc_w[m * 2048 + c] : score_w[(m - 324) * 2048 + c];
    out[(long)m * 2048 + c] = (__bf16)v;
}

// h = relu(col2 + row2), fp32 (490 x 3750)
__global__ void k_combine_relu(const float* __restrict__ a, const float* __restrict__ b,
                               float* __restrict__ h, int n) {
    int i = blockIdx.x * blockDim.x + threadIdx.x;
    if (i >= n) return;
    float v = a[i] + b[i];
    h[i] = v > 0.0f ? v : 0.0f;
}

// PSROI-align max pool: h (490,50,75) fp32 + rois (1000,4) -> pooledB[n][k] bf16
__global__ void k_psroi(const float* __restrict__ h, const float* __restrict__ rois,
                        __bf16* __restrict__ pooled) {
    int i = blockIdx.x * blockDim.x + threadIdx.x;
    if (i >= 1000 * 490) return;
    int r = i / 490, c = i % 490;
    int rem = c % 49;
    int gh = rem / 7, gw = rem % 7;
    const float SC = 1.0f / 16.0f;
    float x1 = rois[r * 4 + 0] * SC, y1 = rois[r * 4 + 1] * SC;
    float x2 = rois[r * 4 + 2] * SC, y2 = rois[r * 4 + 3] * SC;
    float bw = (x2 - x1) / 7.0f, bh = (y2 - y1) / 7.0f;
    const float* f = h + (long)c * 3750;
    float best = -3.4e38f;
#pragma unroll
    for (int sy = 0; sy < 2; ++sy) {
#pragma unroll
        for (int sx = 0; sx < 2; ++sx) {
            float xs = x1 + ((float)gw + ((float)sx + 0.5f) * 0.5f) * bw;
            float ys = y1 + ((float)gh + ((float)sy + 0.5f) * 0.5f) * bh;
            xs = fminf(fmaxf(xs, 0.0f), 74.0f);
            ys = fminf(fmaxf(ys, 0.0f), 49.0f);
            int x0 = (int)floorf(xs), y0 = (int)floorf(ys);
            float lx = xs - (float)x0, ly = ys - (float)y0;
            int x1i = min(x0 + 1, 74), y1i = min(y0 + 1, 49);
            float v00 = f[y0 * 75 + x0],  v01 = f[y0 * 75 + x1i];
            float v10 = f[y1i * 75 + x0], v11 = f[y1i * 75 + x1i];
            float v = (1.0f - ly) * ((1.0f - lx) * v00 + lx * v01)
                    + ly        * ((1.0f - lx) * v10 + lx * v11);
            best = fmaxf(best, v);
        }
    }
    pooled[(long)r * 512 + c] = (__bf16)best;
}

// ---------------------------------------------------------------------------

extern "C" void kernel_launch(void* const* d_in, const int* in_sizes, int n_in,
                              void* d_out, int out_size, void* d_ws, size_t ws_size,
                              hipStream_t stream) {
    (void)in_sizes; (void)n_in; (void)out_size; (void)ws_size;

    const float* x         = (const float*)d_in[0];
    const float* rois      = (const float*)d_in[1];
    const float* w_col_max = (const float*)d_in[2];
    const float* b_col_max = (const float*)d_in[3];
    const float* w_col     = (const float*)d_in[4];
    const float* b_col     = (const float*)d_in[5];
    const float* w_row_max = (const float*)d_in[6];
    const float* b_row_max = (const float*)d_in[7];
    const float* w_row     = (const float*)d_in[8];
    const float* b_row     = (const float*)d_in[9];
    const float* fc1_w     = (const float*)d_in[10];
    const float* fc1_b     = (const float*)d_in[11];
    const float* score_w   = (const float*)d_in[12];
    const float* score_b   = (const float*)d_in[13];
    const float* loc_w     = (const float*)d_in[14];
    const float* loc_b     = (const float*)d_in[15];
    float* out = (float*)d_out;

    // ----- workspace carve-up (all pointer math; memset once per call) -----
    char* ws = (char*)d_ws;
    size_t off = 0;
    auto alloc = [&](size_t bytes) -> void* {
        void* p = ws + off;
        off = (off + bytes + 255) & ~(size_t)255;
        return p;
    };
    __bf16* xpadT   = (__bf16*)alloc((size_t)6144 * 2048 * 2);      // padded x, slack rows
    __bf16* A1col   = (__bf16*)alloc((size_t)15 * 256 * 2048 * 2);
    __bf16* A1row   = (__bf16*)alloc((size_t)15 * 256 * 2048 * 2);
    __bf16* A2col   = (__bf16*)alloc((size_t)15 * 512 * 256 * 2);
    __bf16* A2row   = (__bf16*)alloc((size_t)15 * 512 * 256 * 2);
    __bf16* Afc1    = (__bf16*)alloc((size_t)2048 * 512 * 2);
    __bf16* Ahead   = (__bf16*)alloc((size_t)512 * 2048 * 2);
    __bf16* midColT = (__bf16*)alloc((size_t)4608 * 256 * 2);       // (50,89) w-padded + slack
    __bf16* midRowT = (__bf16*)alloc((size_t)4992 * 256 * 2);       // (64,75) h-padded + slack
    float*  col2    = (float*)alloc((size_t)490 * 3750 * 4);
    float*  row2    = (float*)alloc((size_t)490 * 3750 * 4);
    float*  hbuf    = (float*)alloc((size_t)490 * 3750 * 4);
    __bf16* pooledB = (__bf16*)alloc((size_t)1024 * 512 * 2);
    __bf16* fc1B    = (__bf16*)alloc((size_t)1024 * 2048 * 2);

    hipMemsetAsync(d_ws, 0, off, stream);

    // ----- repack inputs to bf16 GEMM layouts -----
    const int TB = 256;
    k_pack_x<<<(2048 * 3750 + TB - 1) / TB, TB, 0, stream>>>(x, xpadT);
    k_pack_w<<<(256 * 2048 * 15 + TB - 1) / TB, TB, 0, stream>>>(w_col_max, A1col, 256, 256, 2048, 2048, 15);
    k_pack_w<<<(256 * 2048 * 15 + TB - 1) / TB, TB, 0, stream>>>(w_row_max, A1row, 256, 256, 2048, 2048, 15);
    k_pack_w<<<(490 * 256 * 15 + TB - 1) / TB, TB, 0, stream>>>(w_col, A2col, 490, 512, 256, 256, 15);
    k_pack_w<<<(490 * 256 * 15 + TB - 1) / TB, TB, 0, stream>>>(w_row, A2row, 490, 512, 256, 256, 15);
    k_pack_w<<<(2048 * 490 + TB - 1) / TB, TB, 0, stream>>>(fc1_w, Afc1, 2048, 2048, 490, 512, 1);
    k_pack_heads<<<(405 * 2048 + TB - 1) / TB, TB, 0, stream>>>(loc_w, score_w, Ahead);

    const int NB_CONV = (3750 + 127) / 128;   // 30

    // ----- conv1 col (15x1 vertical, pad h=7): 256 x 2048x15 x 3750 -----
    {
        GemmP p{};
        p.A = A1col; p.B = xpadT; p.bias = b_col_max;
        p.M = 256; p.N = 3750; p.Mp = 256; p.Kc = 2048;
        p.taps = 15; p.tapStride = 89; p.bRowStride = 89; p.bColBase = 7; p.nDiv = 75;
        p.outH = midColT; p.oRowStride = 89; p.oBase = 7; p.oCp = 256;
        gemm_bf16_wmma<1><<<dim3(2, NB_CONV), 256, 0, stream>>>(p);
    }
    // ----- conv1 row (1x15 horizontal, pad w=7) -----
    {
        GemmP p{};
        p.A = A1row; p.B = xpadT; p.bias = b_row_max;
        p.M = 256; p.N = 3750; p.Mp = 256; p.Kc = 2048;
        p.taps = 15; p.tapStride = 1; p.bRowStride = 89; p.bColBase = 7 * 89; p.nDiv = 75;
        p.outH = midRowT; p.oRowStride = 75; p.oBase = 525; p.oCp = 256;
        gemm_bf16_wmma<1><<<dim3(2, NB_CONV), 256, 0, stream>>>(p);
    }
    // ----- conv2 col (1x15 horizontal on midCol): 490 x 256x15 x 3750 -----
    {
        GemmP p{};
        p.A = A2col; p.B = midColT; p.bias = b_col;
        p.M = 490; p.N = 3750; p.Mp = 512; p.Kc = 256;
        p.taps = 15; p.tapStride = 1; p.bRowStride = 89; p.bColBase = 0; p.nDiv = 75;
        p.outF = col2; p.ldF = 3750;
        gemm_bf16_wmma<0><<<dim3(4, NB_CONV), 256, 0, stream>>>(p);
    }
    // ----- conv2 row (15x1 vertical on midRow) -----
    {
        GemmP p{};
        p.A = A2row; p.B = midRowT; p.bias = b_row;
        p.M = 490; p.N = 3750; p.Mp = 512; p.Kc = 256;
        p.taps = 15; p.tapStride = 75; p.bRowStride = 75; p.bColBase = 0; p.nDiv = 75;
        p.outF = row2; p.ldF = 3750;
        gemm_bf16_wmma<0><<<dim3(4, NB_CONV), 256, 0, stream>>>(p);
    }

    // ----- h = relu(col2 + row2); PSROI max pool -----
    k_combine_relu<<<(490 * 3750 + TB - 1) / TB, TB, 0, stream>>>(col2, row2, hbuf, 490 * 3750);
    k_psroi<<<(1000 * 490 + TB - 1) / TB, TB, 0, stream>>>(hbuf, rois, pooledB);

    // ----- fc1: 2048 x 512 x 1024, relu, store [n][m] bf16 -----
    {
        GemmP p{};
        p.A = Afc1; p.B = pooledB; p.bias = fc1_b;
        p.M = 2048; p.N = 1024; p.Mp = 2048; p.Kc = 512;
        p.taps = 1; p.tapStride = 0; p.bRowStride = 0; p.bColBase = 0; p.nDiv = 1 << 30;
        p.outH = fc1B; p.oCp = 2048;
        gemm_bf16_wmma<2><<<dim3(16, 8), 256, 0, stream>>>(p);
    }
    // ----- heads: (324 loc + 81 score) x 2048 x 1000, scatter to d_out -----
    {
        GemmP p{};
        p.A = Ahead; p.B = fc1B; p.bias = nullptr;
        p.M = 405; p.N = 1000; p.Mp = 512; p.Kc = 2048;
        p.taps = 1; p.tapStride = 0; p.bRowStride = 0; p.bColBase = 0; p.nDiv = 1 << 30;
        p.out0 = out; p.bias0 = loc_b; p.M0 = 324;
        p.out1 = out + 1000 * 324; p.bias1 = score_b;
        gemm_bf16_wmma<3><<<dim3(4, 8), 256, 0, stream>>>(p);
    }
}
